// DebertaAttention_21320217658132
// MI455X (gfx1250) — matrix operand order repaired and checked by reading, not compiled, not run on hardware
//
#include <hip/hip_runtime.h>

// ---------------------------------------------------------------------------
// DeBERTa disentangled attention for gfx1250 (MI455X), bf16 WMMA everywhere.
// ---------------------------------------------------------------------------

typedef __attribute__((ext_vector_type(16))) __bf16 v16bf;
typedef __attribute__((ext_vector_type(8)))  __bf16 v8bf;
typedef __attribute__((ext_vector_type(8)))  float  v8f;

#define NH     16
#define DH     64
#define SEQ    1024
#define HID    1024
#define BATCH  2
#define SPAN2  1024               // 2*MAX_REL window rows
#define INV_SCALE 0.07216878365f  // 1/sqrt(64*3)

// ---- WMMA operand loaders (layouts per CDNA5 ISA §7.12.2) ------
// A 16x32 bf16: lane L (m = L&15, hi = L>>4) holds K = c0+8*hi..+7 in halves
// 0..7 and K = c0+16+8*hi..+7 in halves 8..15.
static __device__ __forceinline__ v16bf ld_a16(const __bf16* base, int ld,
                                               int m0, int c0, int lane) {
  int m  = m0 + (lane & 15);
  int hi = lane >> 4;
  const __bf16* p = base + (long)m * ld + c0 + 8 * hi;
  v8bf lo = *(const v8bf*)p;
  v8bf hh = *(const v8bf*)(p + 16);
  v16bf r;
#pragma unroll
  for (int i = 0; i < 8; ++i) { r[i] = lo[i]; r[i + 8] = hh[i]; }
  return r;
}

// B 32x16 bf16 where B_mat[k][n] = Wrow[n][k]: lane L (n = L&15, hi = L>>4)
// holds Wrow[n][c0+8*hi .. +7] in halves 0..7 and Wrow[n][c0+16+8*hi .. +7] in 8..15.
static __device__ __forceinline__ v16bf ld_b16(const __bf16* base, int ld,
                                               int n0, int c0, int lane) {
  int n  = n0 + (lane & 15);
  int hi = lane >> 4;
  const __bf16* p = base + (long)n * ld + c0 + 8 * hi;
  v8bf lo = *(const v8bf*)p;
  v8bf hh = *(const v8bf*)(p + 16);
  v16bf r;
#pragma unroll
  for (int i = 0; i < 8; ++i) { r[i] = lo[i]; r[i + 8] = hh[i]; }
  return r;
}

static __device__ __forceinline__ v8f wmma_bf16(v16bf a, v16bf b, v8f c) {
  return __builtin_amdgcn_wmma_f32_16x16x32_bf16(false, a, false, b,
                                                 (short)0, c, false, false);
}

// ---- elementwise f32 -> bf16 ----------------------------------------------
__global__ __launch_bounds__(256) void f2bf_kernel(const float* __restrict__ x,
                                                   __bf16* __restrict__ y,
                                                   int n) {
  int i = blockIdx.x * 256 + threadIdx.x;
  if (i < n) y[i] = (__bf16)x[i];
}

// ---- generic NT GEMM: C[m,n] = sum_k A[m,k] * W[n,k], 64x64 tile / WG -----
// MODE 0: qkv projection  -> scatter q/k/v (q: +q_bias, *INV_SCALE; v: +v_bias)
// MODE 1: pos_k = rel @ Wpk^T             -> scatter [h][p][d] bf16
// MODE 2: pos_q = (rel @ Wpq^T + b)*scale -> scatter [h][p][d] bf16
// MODE 3: out   = ctx @ Wout^T + b + residual -> f32 row-major
template <int MODE>
__global__ __launch_bounds__(128) void gemm_nt(
    const __bf16* __restrict__ A, const __bf16* __restrict__ W,
    int K, int Ntot,
    const float* __restrict__ bias0, const float* __restrict__ bias1,
    const float* __restrict__ resid, float* __restrict__ outF,
    __bf16* __restrict__ o0, __bf16* __restrict__ o1, __bf16* __restrict__ o2) {
  const int lane = threadIdx.x & 31;
  const int wv   = threadIdx.x >> 5;
  const int n0   = blockIdx.x * 64;
  const int m0   = blockIdx.y * 64 + wv * 16;

  v8f acc[4] = {};
  for (int c0 = 0; c0 < K; c0 += 32) {
    // prefetch next K chunk of the A strip (global_prefetch_b8)
    if (c0 + 32 < K)
      __builtin_prefetch(A + (long)(m0 + (lane & 15)) * K + c0 + 32, 0, 3);
    v16bf a = ld_a16(A, K, m0, c0, lane);
#pragma unroll
    for (int t = 0; t < 4; ++t) {
      v16bf bm = ld_b16(W, K, n0 + 16 * t, c0, lane);
      acc[t] = wmma_bf16(a, bm, acc[t]);
    }
  }

  const int hi = lane >> 4, nl = lane & 15;
#pragma unroll
  for (int t = 0; t < 4; ++t) {
#pragma unroll
    for (int j = 0; j < 8; ++j) {
      const int m = m0 + 8 * hi + j;
      const int n = n0 + 16 * t + nl;
      float v = acc[t][j];
      if (MODE == 0) {
        int hh = n / 192, t3 = (n / 64) % 3, d = n & 63;
        int bb = m >> 10, s = m & 1023;
        long idx = (((long)(bb * NH + hh)) * SEQ + s) * DH + d;
        if (t3 == 0)      o0[idx] = (__bf16)((v + bias0[hh * DH + d]) * INV_SCALE);
        else if (t3 == 1) o1[idx] = (__bf16)v;
        else              o2[idx] = (__bf16)(v + bias1[hh * DH + d]);
      } else if (MODE == 1) {
        int hh = n >> 6, d = n & 63;
        o0[((long)hh * SPAN2 + m) * DH + d] = (__bf16)v;
      } else if (MODE == 2) {
        int hh = n >> 6, d = n & 63;
        o0[((long)hh * SPAN2 + m) * DH + d] = (__bf16)((v + bias0[n]) * INV_SCALE);
      } else {
        outF[(long)m * Ntot + n] = v + bias0[n] + resid[(long)m * Ntot + n];
      }
    }
  }
}

// ---- fused attention with disentangled bias, flash-style online softmax ----
// grid = (S/64, NH, B), block = 128 (4 waves). Wave wv owns q rows 16wv..+15
// (and k rows 16wv..+15 for the p2c GEMM).
static constexpr unsigned ATTN_SMEM = 149248;  // bytes (see layout below)

__global__ __launch_bounds__(128) void attn_kernel(
    const __bf16* __restrict__ qb, const __bf16* __restrict__ kb,
    const __bf16* __restrict__ vb, const __bf16* __restrict__ pk,
    const __bf16* __restrict__ pq, const int* __restrict__ mask,
    __bf16* __restrict__ ctx) {
  extern __shared__ float4 smem_f4[];
  char* smem = (char*)smem_f4;
  __bf16* qs   = (__bf16*)smem;          // 64x64  q tile          ( 8KB)
  __bf16* ks   = qs  + 64 * 64;          // 64x64  k tile          ( 8KB)
  __bf16* vT   = ks  + 64 * 64;          // 64x64  v transposed    ( 8KB)
  __bf16* pkw  = vT  + 64 * 64;          // 128x64 pos_k window    (16KB)
  __bf16* pqw  = pkw + 128 * 64;         // 128x64 pos_q window    (16KB)
  __bf16* Pb   = pqw + 128 * 64;         // 64x64  probs bf16      ( 8KB)
  float*  sS   = (float*)(Pb + 64 * 64); // 64x64  scores f32      (16KB)
  float*  sC   = sS + 64 * 64;           // 64x128 c2p f32         (32KB)
  float*  sP   = sC + 64 * 128;          // 64x128 p2c f32         (32KB)
  float*  rowm = sP + 64 * 128;          // 64 running max
  float*  rowl = rowm + 64;              // 64 running sum
  float*  rowc = rowl + 64;              // 64 correction
  float*  pm   = rowc + 64;              // 128 partial max
  float*  ps   = pm + 128;               // 128 partial sum

  const int tid  = threadIdx.x;
  const int lane = tid & 31;
  const int wv   = tid >> 5;
  const int q0   = blockIdx.x * 64;
  const int h    = blockIdx.y;
  const int b    = blockIdx.z;
  const int m0   = wv * 16;
  const int hi   = lane >> 4, nl = lane & 15;

  const __bf16* Q  = qb + ((long)(b * NH + h)) * SEQ * DH;
  const __bf16* K  = kb + ((long)(b * NH + h)) * SEQ * DH;
  const __bf16* V  = vb + ((long)(b * NH + h)) * SEQ * DH;
  const __bf16* PK = pk + (long)h * SPAN2 * DH;
  const __bf16* PQ = pq + (long)h * SPAN2 * DH;

  { // q tile: thread t loads 32 elems of row t/2
    int r = tid >> 1, cg = (tid & 1) * 32;
    const float4* src = (const float4*)(Q + (long)(q0 + r) * DH + cg);
    float4* dst = (float4*)(qs + r * 64 + cg);
#pragma unroll
    for (int i = 0; i < 4; ++i) dst[i] = src[i];
  }
  if (tid < 64) { rowm[tid] = -3.0e38f; rowl[tid] = 0.f; }
  v8f Oacc[4] = {};
  __syncthreads();

  for (int k0 = 0; k0 < SEQ; k0 += 64) {
    { // k tile + transposed v tile
      int r = tid >> 1, cg = (tid & 1) * 32;
      const float4* src = (const float4*)(K + (long)(k0 + r) * DH + cg);
      float4* dst = (float4*)(ks + r * 64 + cg);
#pragma unroll
      for (int i = 0; i < 4; ++i) dst[i] = src[i];
      const __bf16* vs = V + (long)(k0 + r) * DH + cg;
#pragma unroll
      for (int c = 0; c < 32; ++c) vT[(cg + c) * 64 + r] = vs[c];
      // prefetch next tile's K/V rows into cache (global_prefetch_b8)
      if (k0 + 64 < SEQ) {
        __builtin_prefetch(K + (long)(k0 + 64 + r) * DH + cg, 0, 3);
        __builtin_prefetch(V + (long)(k0 + 64 + r) * DH + cg, 0, 3);
      }
    }
    // relative-position windows; clamped load absorbs the clip() exactly.
    // idx(q,k) = clip(q-k+512) = clip(Wb + w), w = 64 + qi - ki in [1,127]
    const int Wb = q0 - k0 + 512 - 64;
    {
      int wr = tid;  // 0..127
      int sr = Wb + wr;
      sr = sr < 0 ? 0 : (sr > SPAN2 - 1 ? SPAN2 - 1 : sr);
      const float4* s1 = (const float4*)(PK + (long)sr * DH);
      float4* d1 = (float4*)(pkw + wr * 64);
#pragma unroll
      for (int i = 0; i < 8; ++i) d1[i] = s1[i];
      const float4* s2 = (const float4*)(PQ + (long)sr * DH);
      float4* d2 = (float4*)(pqw + wr * 64);
#pragma unroll
      for (int i = 0; i < 8; ++i) d2[i] = s2[i];
      // prefetch next tile's window rows (base shifts by -64)
      if (k0 + 64 < SEQ) {
        int sn = Wb - 64 + wr;
        sn = sn < 0 ? 0 : (sn > SPAN2 - 1 ? SPAN2 - 1 : sn);
        __builtin_prefetch(PK + (long)sn * DH, 0, 3);
        __builtin_prefetch(PQ + (long)sn * DH, 0, 3);
      }
    }
    __syncthreads();

    // QK^T strip (16x64)
#pragma unroll
    for (int t = 0; t < 4; ++t) {
      v8f acc = {};
#pragma unroll
      for (int c0 = 0; c0 < 64; c0 += 32) {
        v16bf a  = ld_a16(qs, 64, m0, c0, lane);
        v16bf bm = ld_b16(ks, 64, 16 * t, c0, lane);
        acc = wmma_bf16(a, bm, acc);
      }
#pragma unroll
      for (int j = 0; j < 8; ++j)
        sS[(m0 + 8 * hi + j) * 64 + 16 * t + nl] = acc[j];
    }
    // c2p strip: q_strip @ pos_k_window^T (16x128)
#pragma unroll
    for (int t = 0; t < 8; ++t) {
      v8f acc = {};
#pragma unroll
      for (int c0 = 0; c0 < 64; c0 += 32) {
        v16bf a  = ld_a16(qs, 64, m0, c0, lane);
        v16bf bm = ld_b16(pkw, 64, 16 * t, c0, lane);
        acc = wmma_bf16(a, bm, acc);
      }
#pragma unroll
      for (int j = 0; j < 8; ++j)
        sC[(m0 + 8 * hi + j) * 128 + 16 * t + nl] = acc[j];
    }
    // p2c strip: k_strip @ pos_q_window^T (16x128)
#pragma unroll
    for (int t = 0; t < 8; ++t) {
      v8f acc = {};
#pragma unroll
      for (int c0 = 0; c0 < 64; c0 += 32) {
        v16bf a  = ld_a16(ks, 64, m0, c0, lane);
        v16bf bm = ld_b16(pqw, 64, 16 * t, c0, lane);
        acc = wmma_bf16(a, bm, acc);
      }
#pragma unroll
      for (int j = 0; j < 8; ++j)
        sP[(m0 + 8 * hi + j) * 128 + 16 * t + nl] = acc[j];
    }
    __syncthreads();

    // combine scores + diagonal gathers + mask; 2 threads per row
    {
      int r = tid >> 1, cb = (tid & 1) * 32;
      const int* mr = mask + ((long)b * SEQ + (q0 + r)) * SEQ + k0 + cb;
      float* srow = sS + r * 64 + cb;
      const float* crow = sC + r * 128;
      float mx = -3.0e38f;
#pragma unroll
      for (int c = 0; c < 32; ++c) {
        int ki = cb + c;
        float v = srow[c] + crow[64 + r - ki] + sP[ki * 128 + 64 + r - ki];
        if (mr[c] == 0) v = -3.0e38f;
        srow[c] = v;
        mx = fmaxf(mx, v);
      }
      pm[tid] = mx;
    }
    __syncthreads();
    {
      int r = tid >> 1, cb = (tid & 1) * 32;
      float m_new = fmaxf(rowm[r], fmaxf(pm[2 * r], pm[2 * r + 1]));
      float* srow = sS + r * 64 + cb;
      float sum = 0.f;
#pragma unroll
      for (int c = 0; c < 32; ++c) {
        float e = __expf(srow[c] - m_new);
        sum += e;
        Pb[r * 64 + cb + c] = (__bf16)e;
      }
      ps[tid] = sum;
    }
    __syncthreads();
    if ((tid & 1) == 0) {
      int r = tid >> 1;
      float m_new = fmaxf(rowm[r], fmaxf(pm[2 * r], pm[2 * r + 1]));
      float corr = __expf(rowm[r] - m_new);
      rowc[r] = corr;
      rowl[r] = rowl[r] * corr + ps[tid] + ps[tid + 1];
      rowm[r] = m_new;
    }
    __syncthreads();

    // rescale running output and accumulate P @ V
#pragma unroll
    for (int t = 0; t < 4; ++t) {
#pragma unroll
      for (int j = 0; j < 8; ++j) Oacc[t][j] *= rowc[m0 + 8 * hi + j];
    }
#pragma unroll
    for (int t = 0; t < 4; ++t) {
#pragma unroll
      for (int c0 = 0; c0 < 64; c0 += 32) {
        v16bf a  = ld_a16(Pb, 64, m0, c0, lane);
        v16bf bm = ld_b16(vT, 64, 16 * t, c0, lane);
        Oacc[t] = wmma_bf16(a, bm, Oacc[t]);
      }
    }
    __syncthreads();
  }

  // normalize + write ctx in (B,S,H) layout with column h*DH+d
#pragma unroll
  for (int t = 0; t < 4; ++t) {
#pragma unroll
    for (int j = 0; j < 8; ++j) {
      int qi = m0 + 8 * hi + j;
      float o = Oacc[t][j] / rowl[qi];
      long row = (long)b * SEQ + q0 + qi;
      ctx[row * HID + h * DH + 16 * t + nl] = (__bf16)o;
    }
  }
}

// ---- LayerNorm over rows of H=1024 ----------------------------------------
__global__ __launch_bounds__(256) void ln_kernel(const float* __restrict__ x,
                                                 const float* __restrict__ w,
                                                 const float* __restrict__ bb,
                                                 float* __restrict__ y) {
  __shared__ float red[34];
  int row = blockIdx.x;
  const float* xr = x + (long)row * HID;
  float s = 0.f, s2 = 0.f;
  for (int i = threadIdx.x; i < HID; i += 256) {
    float v = xr[i];
    s += v; s2 += v * v;
  }
  for (int o = 16; o > 0; o >>= 1) {
    s  += __shfl_xor(s, o, 32);
    s2 += __shfl_xor(s2, o, 32);
  }
  int lane = threadIdx.x & 31, wv = threadIdx.x >> 5;
  if (lane == 0) { red[wv] = s; red[8 + wv] = s2; }
  __syncthreads();
  if (threadIdx.x == 0) {
    float a = 0.f, c = 0.f;
    for (int i = 0; i < 8; ++i) { a += red[i]; c += red[8 + i]; }
    red[32] = a; red[33] = c;
  }
  __syncthreads();
  float mean = red[32] * (1.0f / HID);
  float var  = red[33] * (1.0f / HID) - mean * mean;
  float inv  = rsqrtf(var + 1e-12f);
  float* yr = y + (long)row * HID;
  for (int i = threadIdx.x; i < HID; i += 256)
    yr[i] = w[i] * ((xr[i] - mean) * inv) + bb[i];
}

// ---------------------------------------------------------------------------
extern "C" void kernel_launch(void* const* d_in, const int* in_sizes, int n_in,
                              void* d_out, int out_size, void* d_ws,
                              size_t ws_size, hipStream_t stream) {
  (void)in_sizes; (void)n_in; (void)out_size; (void)ws_size;
  const float* hs    = (const float*)d_in[0];
  const int*   mask  = (const int*)d_in[1];
  const float* rel   = (const float*)d_in[2];
  const float* wqkv  = (const float*)d_in[3];
  const float* qbias = (const float*)d_in[4];
  const float* vbias = (const float*)d_in[5];
  const float* wpk   = (const float*)d_in[6];
  const float* wpq   = (const float*)d_in[7];
  const float* bpq   = (const float*)d_in[8];
  const float* wout  = (const float*)d_in[9];
  const float* bout  = (const float*)d_in[10];
  const float* lnw   = (const float*)d_in[11];
  const float* lnb   = (const float*)d_in[12];
  float* out = (float*)d_out;

  char* w = (char*)d_ws;
  const size_t MB = 1024 * 1024;
  __bf16* hid_bf  = (__bf16*)(w + 0 * MB);   // 4 MB  (2048x1024)
  __bf16* wqkv_bf = (__bf16*)(w + 4 * MB);   // 6 MB  (3072x1024)
  __bf16* rel_bf  = (__bf16*)(w + 10 * MB);  // 2 MB  (1024x1024)
  __bf16* wpk_bf  = (__bf16*)(w + 12 * MB);  // 2 MB
  __bf16* wpq_bf  = (__bf16*)(w + 14 * MB);  // 2 MB
  __bf16* wout_bf = (__bf16*)(w + 16 * MB);  // 2 MB
  __bf16* qbuf    = (__bf16*)(w + 18 * MB);  // 4 MB  [b,h,s,d]
  __bf16* kbuf    = (__bf16*)(w + 22 * MB);  // 4 MB
  __bf16* vbuf    = (__bf16*)(w + 26 * MB);  // 4 MB
  __bf16* pkbuf   = (__bf16*)(w + 30 * MB);  // 2 MB  [h,p,d]
  __bf16* pqbuf   = (__bf16*)(w + 32 * MB);  // 2 MB
  __bf16* ctxb    = (__bf16*)(w + 34 * MB);  // 4 MB  (2048x1024)
  float*  tmpf    = (float*)(w + 38 * MB);   // 8 MB  (2048x1024)

  auto cvt = [&](const float* src, __bf16* dst, int n) {
    f2bf_kernel<<<(n + 255) / 256, 256, 0, stream>>>(src, dst, n);
  };
  cvt(hs,   hid_bf,  BATCH * SEQ * HID);
  cvt(wqkv, wqkv_bf, 3 * HID * HID);
  cvt(rel,  rel_bf,  SPAN2 * HID);
  cvt(wpk,  wpk_bf,  HID * HID);
  cvt(wpq,  wpq_bf,  HID * HID);
  cvt(wout, wout_bf, HID * HID);

  // qkv = hidden @ in_proj^T (+bias scatter)
  gemm_nt<0><<<dim3(48, 32), 128, 0, stream>>>(
      hid_bf, wqkv_bf, HID, 3 * HID, qbias, vbias, nullptr, nullptr,
      qbuf, kbuf, vbuf);
  // pos_k = rel @ pos_proj^T
  gemm_nt<1><<<dim3(16, 16), 128, 0, stream>>>(
      rel_bf, wpk_bf, HID, HID, nullptr, nullptr, nullptr, nullptr,
      pkbuf, nullptr, nullptr);
  // pos_q = (rel @ pos_q_proj^T + b) / scale
  gemm_nt<2><<<dim3(16, 16), 128, 0, stream>>>(
      rel_bf, wpq_bf, HID, HID, bpq, nullptr, nullptr, nullptr,
      pqbuf, nullptr, nullptr);

  // fused attention with c2p/p2c windows + online softmax
  attn_kernel<<<dim3(SEQ / 64, NH, BATCH), 128, ATTN_SMEM, stream>>>(
      qbuf, kbuf, vbuf, pkbuf, pqbuf, mask, ctxb);

  // out = ctx @ out_dense^T + b + residual
  gemm_nt<3><<<dim3(16, 32), 128, 0, stream>>>(
      ctxb, wout_bf, HID, HID, bout, nullptr, hs, tmpf,
      nullptr, nullptr, nullptr);

  ln_kernel<<<BATCH * SEQ, 256, 0, stream>>>(tmpf, lnw, lnb, out);
}
